// PairScoreModel_37512244363487
// MI455X (gfx1250) — matrix operand order repaired and checked by reading, not compiled, not run on hardware
//
#include <hip/hip_runtime.h>

// ---------------- problem constants ----------------
#define B_    8
#define P_    200
#define S_    10000
#define C_    2
#define O1_   128
#define K1_   256
#define ST1_  32
#define L1_   305      // (10000-256)/32+1
#define O2_   16
#define K2_   8
#define ST2_  2
#define L2_   149      // (305-8)/2+1
#define F_    2384     // O2*L2
#define H_    64
#define PPAD  208      // 13*16, padded pop
#define FPAD  2400     // 75*32, padded feature K
#define KC1   512      // C_*K1_  (conv1 reduction)
#define KC2   1024     // O1_*K2_ (conv2 reduction)
#define K3_   4768     // 2F = 149*32 exactly

typedef __attribute__((ext_vector_type(16))) __bf16 v16bf;
typedef __attribute__((ext_vector_type(8)))  float  v8f;

static __device__ __forceinline__ v8f wmma_bf16(v16bf a, v16bf b, v8f c) {
  return __builtin_amdgcn_wmma_f32_16x16x32_bf16(false, a, false, b, (short)0, c,
                                                 false, false);
}

// A fragment (16x32 bf16), row-major source, row stride `ld` elements.
// Lane L: row m = m0 + (L&15); half = L>>4.
// element e -> k = half*16 + e   (contiguous order, matches B fragments)
static __device__ __forceinline__ v16bf frag_a(const __bf16* src, int ld, int m0,
                                               int k0) {
  const int lane = threadIdx.x & 31;
  const int half = lane >> 4;
  const __bf16* p = src + (size_t)(m0 + (lane & 15)) * ld + k0;
  v16bf a;
#pragma unroll
  for (int e = 0; e < 16; ++e) {
    const int k = half * 16 + e;
    a[e] = p[k];
  }
  return a;
}

// B fragment (32x16 bf16), source row-major [K][N] with row stride ld.
// Lane L: col n = n0 + (L&15); element e -> k = (L>>4)*16 + e.
static __device__ __forceinline__ v16bf frag_b(const __bf16* src, int ld, int k0,
                                               int n0) {
  const int lane = threadIdx.x & 31;
  const int half = lane >> 4;
  const __bf16* p = src + (size_t)(k0 + half * 16) * ld + (n0 + (lane & 15));
  v16bf b;
#pragma unroll
  for (int e = 0; e < 16; ++e) b[e] = p[(size_t)e * ld];
  return b;
}

// B fragment where B[k][n] = src[n*ld + k] (i.e. B = src^T): contiguous loads.
static __device__ __forceinline__ v16bf frag_bT(const __bf16* src, int ld, int k0,
                                                int n0) {
  const int lane = threadIdx.x & 31;
  const int half = lane >> 4;
  const __bf16* p = src + (size_t)(n0 + (lane & 15)) * ld + k0 + half * 16;
  v16bf b;
#pragma unroll
  for (int e = 0; e < 16; ++e) b[e] = p[e];
  return b;
}

// ---------------- weight prep ----------------
// Wc1[j][o], j = tap*2 + c  (matches contiguous x window order), 512x128
__global__ void prep_w1(const float* __restrict__ W1, __bf16* __restrict__ Wc1) {
  int i = blockIdx.x * blockDim.x + threadIdx.x;
  if (i >= KC1 * O1_) return;
  int j = i / O1_, o = i % O1_;
  int c = j & 1, tap = j >> 1;
  Wc1[i] = (__bf16)W1[(o * C_ + c) * K1_ + tap];
}
// Wc2[j][o2], j = tap*128 + o1 (matches contiguous h1 window order), 1024x16
__global__ void prep_w2(const float* __restrict__ W2, __bf16* __restrict__ Wc2) {
  int i = blockIdx.x * blockDim.x + threadIdx.x;
  if (i >= KC2 * O2_) return;
  int j = i / O2_, o2 = i % O2_;
  int tap = j / O1_, o1 = j % O1_;
  Wc2[i] = (__bf16)W2[(o2 * O1_ + o1) * K2_ + tap];
}
// Wc3[k][h] = W3[k][h] (already K-major), 4768x64
__global__ void prep_w3(const float* __restrict__ W3, __bf16* __restrict__ Wc3) {
  int i = blockIdx.x * blockDim.x + threadIdx.x;
  if (i < K3_ * H_) Wc3[i] = (__bf16)W3[i];
}
__global__ void zero_bf16(__bf16* __restrict__ p, int n) {
  int i = blockIdx.x * blockDim.x + threadIdx.x;
  if (i < n) p[i] = (__bf16)0.f;
}

// ---------------- fused conv1+conv2 per individual ----------------
// grid = B*P blocks, 256 threads (8 waves).
// LDS: [ xs bf16 41984B | region shared by { xsf f32 80000B , h1 bf16 83456B } ]
// xsf is only live between async staging and the bf16 conversion barrier;
// h1 is only written after that barrier, so they can alias.
__global__ __launch_bounds__(256) void conv_feats(
    const float* __restrict__ x, const __bf16* __restrict__ Wc1,
    const float* __restrict__ b1, const __bf16* __restrict__ Wc2,
    const float* __restrict__ b2, __bf16* __restrict__ featsbf) {
  __shared__ __align__(16) char smem[41984 + 83456];
  __bf16* xs  = (__bf16*)smem;           // 20992 bf16 (20000 + M-tile overrun pad)
  float*  xsf = (float*)(smem + 41984);  // 20000 f32 staging (async target)
  __bf16* h1  = (__bf16*)(smem + 41984); // 41728 bf16, rows 0..325 reachable
  const int bp = blockIdx.x;
  const int b = bp / P_, p = bp % P_;
  const int tid = threadIdx.x;
  const int wave = tid >> 5, lane = tid & 31;

  // ---- async-stage x slice (f32) straight into LDS (ASYNCcnt-tracked) ----
  const float* xg = x + (size_t)bp * (S_ * C_);
  for (int i = tid; i < (S_ * C_) / 4; i += 256) {        // 5000 x 16B chunks
    unsigned ldsa = (unsigned)(size_t)(xsf + i * 4);
    unsigned long long ga = (unsigned long long)(size_t)(xg + i * 4);
    asm volatile("global_load_async_to_lds_b128 %0, %1, off"
                 :: "v"(ldsa), "v"(ga) : "memory");
  }
  // zero pads that do not alias xsf: xs tail, and h1 rows 320..325
  for (int i = S_ * C_ + tid; i < 20992; i += 256) xs[i] = (__bf16)0.f;
  for (int i = 320 * O1_ + tid; i < 41728; i += 256) h1[i] = (__bf16)0.f;
  asm volatile("s_wait_asynccnt 0x0" ::: "memory");
  __syncthreads();

  // ---- convert staged f32 -> bf16 im2col buffer ----
  for (int i = tid; i < S_ * C_; i += 256) xs[i] = (__bf16)xsf[i];
  __syncthreads();   // xsf dead from here; h1 may now be written

  // conv1 implicit GEMM: M=305(20 tiles) N=128 K=512. wave w owns N-tile w.
  {
    const int n0 = wave * 16;
    v16bf bw[16];
#pragma unroll
    for (int kk = 0; kk < 16; ++kk) bw[kk] = frag_b(Wc1, O1_, kk * 32, n0);
    const float bias = b1[n0 + (lane & 15)];
    for (int mt = 0; mt < 20; ++mt) {
      v8f acc = {};
#pragma unroll
      for (int kk = 0; kk < 16; ++kk)   // im2col row stride = 64 bf16
        acc = wmma_bf16(frag_a(xs, 64, mt * 16, kk * 32), bw[kk], acc);
#pragma unroll
      for (int r = 0; r < 8; ++r) {
        int m = mt * 16 + r + 8 * (lane >> 4);
        float v = acc[r] + bias;
        h1[m * O1_ + n0 + (lane & 15)] = (__bf16)(v > 0.f ? v : 0.f);
      }
    }
  }
  __syncthreads();

  // conv2 implicit GEMM: M=149(10 tiles) N=16 K=1024. im2col row stride = 256.
  for (int mt = wave; mt < 10; mt += 8) {
    v8f acc = {};
    for (int kk = 0; kk < 32; ++kk)
      acc = wmma_bf16(frag_a(h1, 256, mt * 16, kk * 32),
                      frag_b(Wc2, O2_, kk * 32, 0), acc);
    const int n = lane & 15;           // o2 channel
    const float bias = b2[n];
#pragma unroll
    for (int r = 0; r < 8; ++r) {
      int m = mt * 16 + r + 8 * (lane >> 4);   // l2 position
      if (m < L2_) {
        float v = acc[r] + bias;                // no relu after conv2
        int fi = n * L2_ + m;                   // reshape (O2, L2) -> F
        featsbf[((size_t)b * PPAD + p) * FPAD + fi] = (__bf16)v;
      }
    }
  }
}

// ---------------- pairwise scores: feats @ feats^T per batch ----------------
// grid = B*13*13 blocks of 32 threads; K padded to 2400.
__global__ __launch_bounds__(32) void pair_scores_k(
    const __bf16* __restrict__ featsbf, float* __restrict__ out) {
  int t = blockIdx.x;
  const int nt = t % 13; t /= 13;
  const int mt = t % 13; const int b = t / 13;
  const __bf16* base = featsbf + (size_t)b * PPAD * FPAD;
  v8f acc = {};
  for (int kk = 0; kk < FPAD / 32; ++kk)
    acc = wmma_bf16(frag_a(base, FPAD, mt * 16, kk * 32),
                    frag_bT(base, FPAD, kk * 32, nt * 16), acc);
  const int lane = threadIdx.x & 31;
  const int n = nt * 16 + (lane & 15);
#pragma unroll
  for (int r = 0; r < 8; ++r) {
    int m = mt * 16 + r + 8 * (lane >> 4);
    if (m < P_ && n < P_)
      out[(size_t)b * (P_ * P_) + (size_t)m * P_ + n] = acc[r];
  }
}

// ---------------- top-k (k = 200) incremental argmax ----------------
// Per-thread segment best cached in LDS; only the winning segment rescans.
__global__ __launch_bounds__(256) void topk_k(const float* __restrict__ scores,
                                              int* __restrict__ top_idx) {
  const int b = blockIdx.x;
  const float* s = scores + (size_t)b * (P_ * P_);
  __shared__ float cv[256]; __shared__ int ci[256];   // persistent candidates
  __shared__ float rv[256]; __shared__ int ri[256];   // reduction buffer
  __shared__ unsigned usedbits[(P_ * P_ + 31) / 32];  // 1250 words
  const int tid = threadIdx.x;
  for (int w = tid; w < (P_ * P_ + 31) / 32; w += 256) usedbits[w] = 0u;
  __syncthreads();
  {
    float bv = -3.402823e38f; int bi = 0x7fffffff;
    for (int i = tid; i < P_ * P_; i += 256) {
      float v = s[i];
      if (v > bv) { bv = v; bi = i; }   // ascending scan -> lowest idx on tie
    }
    cv[tid] = bv; ci[tid] = bi;
  }
  __syncthreads();
  for (int k = 0; k < P_; ++k) {
    rv[tid] = cv[tid]; ri[tid] = ci[tid];
    __syncthreads();
    for (int off = 128; off > 0; off >>= 1) {
      if (tid < off) {
        float ov = rv[tid + off]; int oi = ri[tid + off];
        if (ov > rv[tid] || (ov == rv[tid] && oi < ri[tid])) {
          rv[tid] = ov; ri[tid] = oi;
        }
      }
      __syncthreads();
    }
    const int win = ri[0];
    if (tid == 0) {
      top_idx[b * P_ + k] = win;
      usedbits[win >> 5] |= (1u << (win & 31));
    }
    __syncthreads();
    if ((win & 255) == tid) {   // owner of winning segment rescans it
      float bv = -3.402823e38f; int bi = 0x7fffffff;
      for (int i = tid; i < P_ * P_; i += 256) {
        if (usedbits[i >> 5] & (1u << (i & 31))) continue;
        float v = s[i];
        if (v > bv) { bv = v; bi = i; }
      }
      cv[tid] = bv; ci[tid] = bi;
    }
    __syncthreads();
  }
}

// ---------------- critic MLP on gathered pair features ----------------
// grid = 100 blocks (16 rows each), 128 threads (4 waves; wave w = N-tile w)
__global__ __launch_bounds__(128) void critic_k(
    const __bf16* __restrict__ featsbf, const int* __restrict__ top_idx,
    const __bf16* __restrict__ Wc3, const float* __restrict__ b3,
    const float* __restrict__ W4, const float* __restrict__ b4,
    float* __restrict__ vout) {
  __shared__ float hid[16 * H_];
  const int t = blockIdx.x;
  const int tid = threadIdx.x;
  const int wave = tid >> 5, lane = tid & 31;
  const int half = lane >> 4;
  // per-lane gathered row: concat(feats[b][row], feats[b][col])
  const int gr = t * 16 + (lane & 15);
  const int b = gr / P_;
  const int idx = top_idx[gr];
  const __bf16* pr = featsbf + ((size_t)b * PPAD + idx / P_) * FPAD;
  const __bf16* pc = featsbf + ((size_t)b * PPAD + idx % P_) * FPAD;
  const int n0 = wave * 16;
  v8f acc = {};
  for (int kk = 0; kk < K3_ / 32; ++kk) {
    v16bf a;
#pragma unroll
    for (int e = 0; e < 16; ++e) {
      int k = kk * 32 + half * 16 + e;
      a[e] = (k < F_) ? pr[k] : pc[k - F_];
    }
    acc = wmma_bf16(a, frag_b(Wc3, H_, kk * 32, n0), acc);
  }
  const int n = n0 + (lane & 15);
  const float bias = b3[n];
#pragma unroll
  for (int r = 0; r < 8; ++r) {
    int m = r + 8 * half;
    float v = acc[r] + bias;
    hid[m * H_ + n] = v > 0.f ? v : 0.f;
  }
  __syncthreads();
  if (tid < 16) {
    float v = b4[0];
#pragma unroll 8
    for (int h = 0; h < H_; ++h) v += hid[tid * H_ + h] * W4[h];
    vout[t * 16 + tid] = v > 0.f ? v : 0.f;
  }
}

// ---------------- host launcher ----------------
extern "C" void kernel_launch(void* const* d_in, const int* in_sizes, int n_in,
                              void* d_out, int out_size, void* d_ws,
                              size_t ws_size, hipStream_t stream) {
  (void)in_sizes; (void)n_in; (void)out_size; (void)ws_size;
  const float* x  = (const float*)d_in[0];
  const float* W1 = (const float*)d_in[1];
  const float* b1 = (const float*)d_in[2];
  const float* W2 = (const float*)d_in[3];
  const float* b2 = (const float*)d_in[4];
  const float* W3 = (const float*)d_in[5];
  const float* b3 = (const float*)d_in[6];
  const float* W4 = (const float*)d_in[7];
  const float* b4 = (const float*)d_in[8];
  float* out = (float*)d_out;                     // [B*P*P | B*P]

  char* ws = (char*)d_ws;
  size_t off = 0;
  auto take = [&](size_t bytes) -> void* {
    void* p = (void*)(ws + off);
    off += (bytes + 255) & ~(size_t)255;
    return p;
  };
  __bf16* Wc1     = (__bf16*)take((size_t)KC1 * O1_ * 2);
  __bf16* Wc2     = (__bf16*)take((size_t)KC2 * O2_ * 2);
  __bf16* Wc3     = (__bf16*)take((size_t)K3_ * H_ * 2);
  __bf16* featsbf = (__bf16*)take((size_t)B_ * PPAD * FPAD * 2);
  int*    topidx  = (int*)   take((size_t)B_ * P_ * 4);

  const int nfb = B_ * PPAD * FPAD;
  hipLaunchKernelGGL(prep_w1, dim3((KC1 * O1_ + 255) / 256), dim3(256), 0, stream,
                     W1, Wc1);
  hipLaunchKernelGGL(prep_w2, dim3((KC2 * O2_ + 255) / 256), dim3(256), 0, stream,
                     W2, Wc2);
  hipLaunchKernelGGL(prep_w3, dim3((K3_ * H_ + 255) / 256), dim3(256), 0, stream,
                     W3, Wc3);
  hipLaunchKernelGGL(zero_bf16, dim3((nfb + 255) / 256), dim3(256), 0, stream,
                     featsbf, nfb);
  hipLaunchKernelGGL(conv_feats, dim3(B_ * P_), dim3(256), 0, stream,
                     x, Wc1, b1, Wc2, b2, featsbf);
  hipLaunchKernelGGL(pair_scores_k, dim3(B_ * 13 * 13), dim3(32), 0, stream,
                     featsbf, out);
  hipLaunchKernelGGL(topk_k, dim3(B_), dim3(256), 0, stream, out, topidx);
  hipLaunchKernelGGL(critic_k, dim3((B_ * P_) / 16), dim3(128), 0, stream,
                     featsbf, topidx, Wc3, b3, W4, b4,
                     out + (size_t)B_ * P_ * P_);
}
